// GIKTOld_66752381714706
// MI455X (gfx1250) — matrix operand order repaired and checked by reading, not compiled, not run on hardware
//
#include <hip/hip_runtime.h>
#include <hip/hip_bf16.h>
#include <stdint.h>
#include <stddef.h>

#ifndef __has_builtin
#define __has_builtin(x) 0
#endif
#if __has_builtin(__builtin_amdgcn_wmma_f32_16x16x4_f32)
#define GIKT_F32_WMMA 1
#endif

// Hardware tanh (gfx1250 V_TANH_F32, a TRANS32 op that co-executes with WMMA)
#if __has_builtin(__builtin_amdgcn_tanhf)
#define HTANH(x) __builtin_amdgcn_tanhf(x)
#elif __has_builtin(__builtin_amdgcn_tanh_f32)
#define HTANH(x) __builtin_amdgcn_tanh_f32(x)
#else
#define HTANH(x) tanhf(x)
#endif

#if __has_builtin(__builtin_amdgcn_tensor_load_to_lds)
#define GIKT_TDM 1
#endif

#define NQk   20000
#define NSk   2000
#define Dk    128
#define Bk    32
#define Sk    200
#define NEGC  (-1.0e30f)

typedef __attribute__((ext_vector_type(2)))  float        v2f;
typedef __attribute__((ext_vector_type(8)))  float        v8f;
typedef __attribute__((ext_vector_type(16))) __bf16       v16bf;
typedef __attribute__((ext_vector_type(4)))  unsigned int v4u;
typedef __attribute__((ext_vector_type(4)))  int          v4i;
typedef __attribute__((ext_vector_type(8)))  int          v8i;

__device__ __forceinline__ float sigmf(float x) { return 0.5f + 0.5f * HTANH(0.5f * x); }

// ---------------------------------------------------------------------------
// Phase-1 index construction:
//   n1[j] = q_neighbors[question[j/4], j%4]          j < 25600
//   n2[i] = s_neighbors[n1[i/4], i%4]                i < 102400
// ---------------------------------------------------------------------------
__global__ void k_indices(const int* __restrict__ question,
                          const int* __restrict__ q_nb,
                          const int* __restrict__ s_nb,
                          int* __restrict__ n1, int* __restrict__ n2)
{
    int i = blockIdx.x * blockDim.x + threadIdx.x;
    if (i < 25600) n1[i] = q_nb[question[i >> 2] * 4 + (i & 3)];
    if (i < 102400) {
        int j   = i >> 2;
        int n1v = q_nb[question[j >> 2] * 4 + (j & 3)];
        n2[i]   = s_nb[n1v * 4 + (i & 3)];
    }
}

__global__ void k_gather_e0(const int* __restrict__ question,
                            const float* __restrict__ embq,
                            float* __restrict__ e0)
{
    int i = blockIdx.x * blockDim.x + threadIdx.x;
    if (i >= 6400 * Dk) return;
    int m = i >> 7, d = i & 127;
    e0[i] = embq[(size_t)question[m] * Dk + d];
}

__global__ void k_gather_e1(const int* __restrict__ n1,
                            const float* __restrict__ embc,
                            float* __restrict__ e1)
{
    int i = blockIdx.x * blockDim.x + threadIdx.x;
    if (i >= 25600 * Dk) return;
    int m = i >> 7, d = i & 127;
    e1[i] = embc[(size_t)n1[m] * Dk + d];
}

// Convert agg_w[0..2] and agg_last_w to bf16 (row-major N x K = 128 x 128 each)
__global__ void k_wbf(const float* __restrict__ agg_w,
                      const float* __restrict__ agg_last_w,
                      __bf16* __restrict__ Wb)
{
    int i = blockIdx.x * blockDim.x + threadIdx.x;
    if (i >= 4 * Dk * Dk) return;
    float v = (i < 3 * Dk * Dk) ? agg_w[i] : agg_last_w[i - 3 * Dk * Dk];
    Wb[i] = (__bf16)v;
}

// X[m,:] = bf16( self[m,:] + mean_k child[4m+k,:] )
__global__ void k_prep_generic(const float* __restrict__ selfb,
                               const float* __restrict__ childb,
                               __bf16* __restrict__ X, int M)
{
    int i = blockIdx.x * blockDim.x + threadIdx.x;
    if (i >= M * Dk) return;
    int m = i >> 7, d = i & 127;
    const float* c = childb + (size_t)(m * 4) * Dk + d;
    X[i] = (__bf16)(selfb[i] + 0.25f * (c[0] + c[Dk] + c[2 * Dk] + c[3 * Dk]));
}

// i=0, j=1:  self = e1 buffer, children = embed_question[n2[4m+k]]
__global__ void k_prep_x1(const float* __restrict__ e1,
                          const int* __restrict__ n2,
                          const float* __restrict__ embq,
                          __bf16* __restrict__ X)
{
    int i = blockIdx.x * blockDim.x + threadIdx.x;
    if (i >= 25600 * Dk) return;
    int m = i >> 7, d = i & 127;
    float s = 0.f;
    #pragma unroll
    for (int k = 0; k < 4; ++k) s += embq[(size_t)n2[m * 4 + k] * Dk + d];
    X[i] = (__bf16)(e1[i] + 0.25f * s);
}

// i=0, j=2:  self = embed_question[n2[m]], children = embed_concept[q_nb[n2[m],k]]
__global__ void k_prep_x2(const int* __restrict__ n2,
                          const int* __restrict__ q_nb,
                          const float* __restrict__ embq,
                          const float* __restrict__ embc,
                          __bf16* __restrict__ X)
{
    int i = blockIdx.x * blockDim.x + threadIdx.x;
    if (i >= 102400 * Dk) return;
    int m = i >> 7, d = i & 127;
    int q = n2[m];
    float s = 0.f;
    #pragma unroll
    for (int k = 0; k < 4; ++k) s += embc[(size_t)q_nb[q * 4 + k] * Dk + d];
    X[i] = (__bf16)(embq[(size_t)q * Dk + d] + 0.25f * s);
}

__global__ void k_copy_bf(const float* __restrict__ src, __bf16* __restrict__ X, int n)
{
    int i = blockIdx.x * blockDim.x + threadIdx.x;
    if (i < n) X[i] = (__bf16)src[i];
}

// ---------------------------------------------------------------------------
// Phase-1 GEMM: out(M,128) = tanh( X(M,128)_bf16 @ W^T(128,128)_bf16 + bias )
// The 32KB weight tile is moved into LDS ONCE per workgroup by the Tensor
// Data Mover (wave 0 issues tensor_load_to_lds, completes on s_wait_tensorcnt,
// workgroup barrier publishes it); all 8 waves then build B fragments from
// LDS while A fragments stay in registers.  4 x v_wmma_f32_16x16x32_bf16 per
// 16x16 output tile (K=128); V_TANH_F32 epilogue.
// ---------------------------------------------------------------------------
__global__ void __launch_bounds__(256)
k_gemm_tanh_bf16(const __bf16* __restrict__ X,
                 const __bf16* __restrict__ W,
                 const float* __restrict__ bias,
                 float* __restrict__ out, int M)
{
    __shared__ __bf16 lds_w[Dk * Dk];   // 32 KB weight tile, shared by 8 waves

#ifdef GIKT_TDM
    if (threadIdx.x < 32) {             // one TDM descriptor, issued by wave 0
        unsigned long long ga = (unsigned long long)(uintptr_t)W;
        unsigned int ldsa = (unsigned int)(uintptr_t)(&lds_w[0]);
        v4u g0;
        g0[0] = 1u;                                   // count=1, user mode
        g0[1] = ldsa;                                 // D#.lds_addr
        g0[2] = (unsigned int)(ga & 0xffffffffu);     // D#.global_addr[31:0]
        g0[3] = (unsigned int)((ga >> 32) & 0x01ffffffu) | 0x80000000u; // addr[56:32] | type=2
        v8i g1;
        g1[0] = 0x00010000;       // wg_mask=0, data_size=1 (2B), no flags
        g1[1] = (Dk << 16);       // tensor_dim0 = 128 (bits 79:48, low half)
        g1[2] = (Dk << 16);       // tensor_dim0 hi=0 | tensor_dim1 = 128 (low half)
        g1[3] = (Dk << 16);       // tensor_dim1 hi=0 | tile_dim0 = 128
        g1[4] = Dk;               // tile_dim1 = 128, tile_dim2 = 0
        g1[5] = Dk;               // tensor_dim0_stride = 128 (low 32)
        g1[6] = 0;                // stride hi | tensor_dim1_stride lo
        g1[7] = 0;
        v4i z4 = {0, 0, 0, 0};
#if defined(__clang_major__) && (__clang_major__ >= 23)
        v8i z8 = {0, 0, 0, 0, 0, 0, 0, 0};
        __builtin_amdgcn_tensor_load_to_lds(g0, g1, z4, z4, z8, 0);
#else
        __builtin_amdgcn_tensor_load_to_lds(g0, g1, z4, z4, 0);
#endif
#if __has_builtin(__builtin_amdgcn_s_wait_tensorcnt)
        __builtin_amdgcn_s_wait_tensorcnt(0);
#else
        asm volatile("s_wait_tensorcnt 0x0" ::: "memory");
#endif
    }
#else
    for (int i = threadIdx.x; i < Dk * Dk; i += 256) lds_w[i] = W[i];
#endif
    __syncthreads();

    const int wave = threadIdx.x >> 5;
    const int lane = threadIdx.x & 31;
    const int half = lane >> 4;
    const int l16  = lane & 15;
    const int mbase = blockIdx.x * 128 + wave * 16;
    if (mbase >= M) return;
    const int mrow = mbase + l16;

    v16bf afr[4];
    #pragma unroll
    for (int kk = 0; kk < 4; ++kk) {
        const __bf16* ap = X + (size_t)mrow * Dk + kk * 32;
        #pragma unroll
        for (int e = 0; e < 8; ++e) afr[kk][e] = ap[half * 16 + e];
        #pragma unroll
        for (int e = 0; e < 8; ++e) afr[kk][8 + e] = ap[half * 16 + 8 + e];
    }
    #pragma unroll 1
    for (int nt = 0; nt < 8; ++nt) {
        const int n = nt * 16 + l16;
        v8f acc = {};
        #pragma unroll
        for (int kk = 0; kk < 4; ++kk) {
            v16bf bfr;
            const __bf16* bp = lds_w + n * Dk + kk * 32 + half * 16;
            #pragma unroll
            for (int e = 0; e < 16; ++e) bfr[e] = bp[e];
            acc = __builtin_amdgcn_wmma_f32_16x16x32_bf16(
                      false, afr[kk], false, bfr, (short)0, acc, false, false);
        }
        const float bn = bias[n];
        #pragma unroll
        for (int r = 0; r < 8; ++r)
            out[(size_t)(mbase + r + 8 * half) * Dk + n] = HTANH(acc[r] + bn);
    }
}

__global__ void k_emb_q_rec(const unsigned char* __restrict__ mask,
                            const int* __restrict__ question,
                            const float* __restrict__ agg,
                            const float* __restrict__ embq,
                            float* __restrict__ rec)
{
    int i = blockIdx.x * blockDim.x + threadIdx.x;
    if (i >= 6400 * Dk) return;
    int m = i >> 7, d = i & 127;
    rec[i] = mask[m] ? agg[i] : embq[(size_t)question[m] * Dk + d];
}

__global__ void k_zero(float* __restrict__ a, int na, float* __restrict__ b, int nb)
{
    int i = blockIdx.x * blockDim.x + threadIdx.x;
    if (i < na) a[i] = 0.f;
    if (i < nb) b[i] = 0.f;
}

// ---------------------------------------------------------------------------
// Scan-kernel GEMM helper: C(32,N) = A(32,K)_lds @ W^T(N,K)_global
// 16 waves split the 2*(N/16) output tiles.  fp32 WMMA (16x16x4) keeps the
// GRU recurrence exact; bf16 WMMA fallback if the f32 builtin is absent.
// ---------------------------------------------------------------------------
__device__ __forceinline__
void gemm32(const float* __restrict__ W, const float* __restrict__ A,
            float* __restrict__ Cd, int N, int K, int wave, int lane)
{
    const int half = lane >> 4, l16 = lane & 15;
    const int ntiles = 2 * (N >> 4);
    for (int tile = wave; tile < ntiles; tile += 16) {
        const int mt = tile & 1, nt = tile >> 1;
        const int m = mt * 16 + l16;
        const int n = nt * 16 + l16;
        v8f acc = {};
#ifdef GIKT_F32_WMMA
        for (int k = 0; k < K; k += 4) {
            v2f a, b;
            a.x = A[m * K + k + 2 * half];
            a.y = A[m * K + k + 2 * half + 1];
            b.x = W[(size_t)n * K + k + 2 * half];
            b.y = W[(size_t)n * K + k + 2 * half + 1];
            acc = __builtin_amdgcn_wmma_f32_16x16x4_f32(
                      false, a, false, b, (short)0, acc, false, false);
        }
#else
        for (int k = 0; k < K; k += 32) {
            v16bf a, b;
            const float* ap = A + m * K + k;
            #pragma unroll
            for (int e = 0; e < 8; ++e) a[e] = (__bf16)ap[half * 16 + e];
            #pragma unroll
            for (int e = 0; e < 8; ++e) a[8 + e] = (__bf16)ap[half * 16 + 8 + e];
            const float* bp = W + (size_t)n * K + k + half * 16;
            #pragma unroll
            for (int e = 0; e < 16; ++e) b[e] = (__bf16)bp[e];
            acc = __builtin_amdgcn_wmma_f32_16x16x32_bf16(
                      false, a, false, b, (short)0, acc, false, false);
        }
#endif
        #pragma unroll
        for (int r = 0; r < 8; ++r)
            Cd[(mt * 16 + r + 8 * half) * N + n] = acc[r];
    }
}

// ---------------------------------------------------------------------------
// Sequential scan: one workgroup, 512 threads = 16 wave32.  ~293KB dynamic LDS
// (WGP allows 320KB).  199 steps.
// ---------------------------------------------------------------------------
__global__ void __launch_bounds__(512)
k_scan(const float* __restrict__ emb_q_rec, float* __restrict__ sh,
       const int* __restrict__ question, const int* __restrict__ response,
       const int* __restrict__ qc_idx, const unsigned char* __restrict__ qc_mask,
       const float* __restrict__ embq, const float* __restrict__ embc,
       const float* __restrict__ embcorr,
       const float* __restrict__ w1ih, const float* __restrict__ w1hh,
       const float* __restrict__ b1ih, const float* __restrict__ b1hh,
       const float* __restrict__ w2ih, const float* __restrict__ w2hh,
       const float* __restrict__ b2ih, const float* __restrict__ b2hh,
       const float* __restrict__ query_w, const float* __restrict__ query_b,
       const float* __restrict__ key_w, const float* __restrict__ key_b,
       const float* __restrict__ mlpW_w, const float* __restrict__ mlpW_b,
       const float* __restrict__ h1_init, const float* __restrict__ h2_init,
       float* __restrict__ yhat)
{
    extern __shared__ float smem[];
    float* sh1 = smem;              // 32x128
    float* sh2 = sh1 + 4096;        // 32x128
    float* sA  = sh2 + 4096;        // 32x256 (gemm A staging)
    float* sC  = sA  + 8192;        // 32x384
    float* sC2 = sC  + 12288;       // 32x384
    float* sG2 = sC2 + 12288;       // 32x128
    float* sSC = sG2 + 4096;        // 32x200 scores
    float* sQC = sSC + 6400;        // 160x128 qc rows (b*5+j)
    float* sOG = sQC + 20480;       // 32x55
    float* sKD = sOG + 1760;        // 352
    float* sQD = sKD + 352;         // 160
    int*   sIDX = (int*)(sQD + 160);  // 32x10
    int*   sHV  = sIDX + 320;         // 32x11

    const int tid  = threadIdx.x;
    const int wave = tid >> 5;
    const int lane = tid & 31;

    for (int i = tid; i < Bk * Dk; i += 512) { sh1[i] = h1_init[i]; sh2[i] = h2_init[i]; }
    __syncthreads();

    for (int t = 0; t < Sk - 1; ++t) {
        // ---- stage xcat = [eq, er] (32x256) and qc rows (160x128) ----
        for (int i = tid; i < Bk * 256; i += 512) {
            int b = i >> 8, c = i & 255;
            float v;
            if (c < Dk) {
                v = emb_q_rec[((size_t)b * Sk + t) * Dk + c];
                if (c == 0 && t + 1 < Sk - 1)   // prefetch next step's row
                    __builtin_prefetch(emb_q_rec + ((size_t)b * Sk + t + 1) * Dk, 0, 1);
            } else {
                v = embcorr[(size_t)response[b * Sk + t] * Dk + (c - Dk)];
            }
            sA[i] = v;
        }
        for (int i = tid; i < 160 * Dk; i += 512) {
            int rw = i >> 7, d = i & 127;
            int b = rw / 5, j = rw % 5;
            int q = question[b * Sk + t + 1];
            float v;
            if (j == 0) v = embq[(size_t)q * Dk + d];
            else {
                int ci = qc_idx[q * 4 + (j - 1)];
                v = embc[(size_t)ci * Dk + d] * (qc_mask[q * 4 + (j - 1)] ? 1.f : 0.f);
            }
            sQC[i] = v;
        }
        __syncthreads();

        // ---- GRU1 ----
        gemm32(w1ih, sA,  sC,  384, 256, wave, lane);
        gemm32(w1hh, sh1, sC2, 384, 128, wave, lane);
        __syncthreads();
        for (int i = tid; i < Bk * Dk; i += 512) {
            int b = i >> 7, d = i & 127;
            float ir = sC [b * 384 + d]        + b1ih[d];
            float iz = sC [b * 384 + 128 + d]  + b1ih[128 + d];
            float in = sC [b * 384 + 256 + d]  + b1ih[256 + d];
            float hr = sC2[b * 384 + d]        + b1hh[d];
            float hz = sC2[b * 384 + 128 + d]  + b1hh[128 + d];
            float hn = sC2[b * 384 + 256 + d]  + b1hh[256 + d];
            float r = sigmf(ir + hr), z = sigmf(iz + hz);
            sh1[i] = (1.f - z) * HTANH(in + r * hn) + z * sh1[i];
        }
        __syncthreads();

        // ---- GRU2 (x = new h1) ----
        gemm32(w2ih, sh1, sC,  384, 128, wave, lane);
        gemm32(w2hh, sh2, sC2, 384, 128, wave, lane);
        __syncthreads();
        for (int i = tid; i < Bk * Dk; i += 512) {
            int b = i >> 7, d = i & 127;
            float ir = sC [b * 384 + d]        + b2ih[d];
            float iz = sC [b * 384 + 128 + d]  + b2ih[128 + d];
            float in = sC [b * 384 + 256 + d]  + b2ih[256 + d];
            float hr = sC2[b * 384 + d]        + b2hh[d];
            float hz = sC2[b * 384 + 128 + d]  + b2hh[128 + d];
            float hn = sC2[b * 384 + 256 + d]  + b2hh[256 + d];
            float r = sigmf(ir + hr), z = sigmf(iz + hz);
            sG2[i] = (1.f - z) * HTANH(in + r * hn) + z * sh2[i];
        }
        // ---- scores sc[b,s] = (s<t) ? <K_hist[b,s], qn[b]> : NEG ----
        for (int i = tid; i < Bk * Sk; i += 512) {
            int b = i / Sk, s = i % Sk;
            float v = NEGC;
            if (s < t) {
                const float* kr = embq + (size_t)question[b * Sk + s] * Dk;
                const float* qr = embq + (size_t)question[b * Sk + t + 1] * Dk;
                float acc = 0.f;
                for (int d = 0; d < Dk; ++d) acc += kr[d] * qr[d];
                v = acc;
            }
            sSC[i] = v;
        }
        __syncthreads();

        // ---- top-10 per row (ties -> lower index), 2 rows per wave ----
        for (int rb = wave; rb < Bk; rb += 16) {
            int chosen[10];
            #pragma unroll
            for (int kk = 0; kk < 10; ++kk) {
                float bv = -__builtin_inff(); int bi = 0x7fffffff;
                for (int s = lane; s < Sk; s += 32) {
                    bool skip = false;
                    #pragma unroll
                    for (int c = 0; c < 10; ++c) if (c < kk && chosen[c] == s) skip = true;
                    if (!skip) {
                        float v = sSC[rb * Sk + s];
                        if (v > bv || (v == bv && s < bi)) { bv = v; bi = s; }
                    }
                }
                for (int off = 16; off; off >>= 1) {
                    float ov = __shfl_xor(bv, off, 32);
                    int   oi = __shfl_xor(bi, off, 32);
                    if (ov > bv || (ov == bv && oi < bi)) { bv = ov; bi = oi; }
                }
                chosen[kk] = bi;
                if (lane == 0) sIDX[rb * 10 + kk] = bi;
            }
            if (lane == 0) sHV[rb * 11] = 1;
            if (lane >= 1 && lane < 11) {
                int cv = 0;
                #pragma unroll
                for (int c = 0; c < 10; ++c) if (lane - 1 == c) cv = chosen[c];
                sHV[rb * 11 + lane] = (cv < t) ? 1 : 0;
            }
        }
        __syncthreads();

        // ---- og[b,q,s] = <qc[b,q], hist[b,s]>   hist[b,0]=g2, else sh[b,idx] ----
        for (int i = tid; i < Bk * 55; i += 512) {
            int b = i / 55, qs = i % 55, q = qs / 11, s = qs % 11;
            const float* qcr = sQC + (b * 5 + q) * Dk;
            const float* hr = (s == 0) ? (sG2 + b * Dk)
                                       : (sh + ((size_t)b * Sk + sIDX[b * 10 + s - 1]) * Dk);
            float acc = 0.f;
            for (int d = 0; d < Dk; ++d) acc += qcr[d] * hr[d];
            sOG[i] = acc;
        }
        __syncthreads();

        // ---- Qp: 5 chunks of 32 qc rows; qdot = <tanh(qc@key_w^T+key_b), w_q> ----
        for (int c = 0; c < 5; ++c) {
            for (int i = tid; i < Bk * Dk; i += 512) sA[i] = sQC[c * Bk * Dk + i];
            __syncthreads();
            gemm32(key_w, sA, sC, 128, 128, wave, lane);
            __syncthreads();
            {
                int r = tid >> 4, lg = tid & 15;
                float part = 0.f;
                for (int d = lg * 8; d < lg * 8 + 8; ++d)
                    part += HTANH(sC[r * Dk + d] + key_b[d]) * mlpW_w[d];
                for (int off = 8; off; off >>= 1) part += __shfl_xor(part, off, 32);
                if (lg == 0) sQD[c * 32 + r] = part;
            }
            __syncthreads();
        }
        // ---- Kp: 11 chunks of 32 hist rows (m=b*11+s); kdot uses query_w, w_k ----
        for (int c = 0; c < 11; ++c) {
            for (int i = tid; i < Bk * Dk; i += 512) {
                int r = i >> 7, d = i & 127;
                int m = c * 32 + r, b = m / 11, s = m % 11;
                sA[i] = (s == 0) ? sG2[b * Dk + d]
                                 : sh[((size_t)b * Sk + sIDX[b * 10 + s - 1]) * Dk + d];
            }
            __syncthreads();
            gemm32(query_w, sA, sC, 128, 128, wave, lane);
            __syncthreads();
            {
                int r = tid >> 4, lg = tid & 15;
                float part = 0.f;
                for (int d = lg * 8; d < lg * 8 + 8; ++d)
                    part += HTANH(sC[r * Dk + d] + query_b[d]) * mlpW_w[Dk + d];
                for (int off = 8; off; off >>= 1) part += __shfl_xor(part, off, 32);
                if (lg == 0) sKD[c * 32 + r] = part;
            }
            __syncthreads();
        }

        // ---- masked softmax over 11 hist slots, p[b] = sum alpha*og ----
        if (tid < Bk) {
            int b = tid;
            float mb = mlpW_b[0], p = 0.f;
            for (int q = 0; q < 5; ++q) {
                float qd = sQD[b * 5 + q], mx = -__builtin_inff();
                float tv[11];
                for (int s = 0; s < 11; ++s) {
                    float v = sHV[b * 11 + s] ? (qd + sKD[b * 11 + s] + mb) : NEGC;
                    tv[s] = v; if (v > mx) mx = v;
                }
                float den = 0.f, num = 0.f;
                for (int s = 0; s < 11; ++s) {
                    float e = __expf(tv[s] - mx);
                    den += e;
                    num += e * sOG[b * 55 + q * 11 + s];
                }
                p += num / den;
            }
            yhat[b * Sk + ((t == 0) ? 0 : t + 1)] = p;
        }
        // ---- state update: h2 = (t==0)? h2 : g2; sh[:,t] = g2 * (t>0) ----
        for (int i = tid; i < Bk * Dk; i += 512) {
            int b = i >> 7, d = i & 127;
            float g = sG2[i];
            if (t > 0) sh2[i] = g;
            sh[((size_t)b * Sk + t) * Dk + d] = (t > 0) ? g : 0.f;
        }
        __syncthreads();
    }
}

// ---------------------------------------------------------------------------
extern "C" void kernel_launch(void* const* d_in, const int* in_sizes, int n_in,
                              void* d_out, int out_size, void* d_ws, size_t ws_size,
                              hipStream_t stream)
{
    const int*  question = (const int*)d_in[0];
    const int*  response = (const int*)d_in[1];
    const unsigned char* mask = (const unsigned char*)d_in[2];
    const int*  q_nb   = (const int*)d_in[3];
    const int*  s_nb   = (const int*)d_in[4];
    const int*  qc_idx = (const int*)d_in[5];
    const unsigned char* qc_mask = (const unsigned char*)d_in[6];
    const float* embq    = (const float*)d_in[7];
    const float* embc    = (const float*)d_in[8];
    const float* embcorr = (const float*)d_in[9];
    const float* w1ih = (const float*)d_in[10];
    const float* w1hh = (const float*)d_in[11];
    const float* b1ih = (const float*)d_in[12];
    const float* b1hh = (const float*)d_in[13];
    const float* w2ih = (const float*)d_in[14];
    const float* w2hh = (const float*)d_in[15];
    const float* b2ih = (const float*)d_in[16];
    const float* b2hh = (const float*)d_in[17];
    const float* agg_w      = (const float*)d_in[18];
    const float* agg_b      = (const float*)d_in[19];
    const float* agg_last_w = (const float*)d_in[20];
    const float* agg_last_b = (const float*)d_in[21];
    const float* query_w = (const float*)d_in[22];
    const float* query_b = (const float*)d_in[23];
    const float* key_w   = (const float*)d_in[24];
    const float* key_b   = (const float*)d_in[25];
    const float* mlpW_w  = (const float*)d_in[26];
    const float* mlpW_b  = (const float*)d_in[27];
    const float* h1_init = (const float*)d_in[28];
    const float* h2_init = (const float*)d_in[29];
    float* yhat = (float*)d_out;

    // ---- workspace carve (~102 MB); phase-1 working set fits in 192MB L2 ----
    char* wsp = (char*)d_ws;
    auto carve = [&](size_t bytes) { char* p = wsp; wsp += (bytes + 255) & ~(size_t)255; return p; };
    int*    n1  = (int*)carve(25600u * 4);
    int*    n2  = (int*)carve(102400u * 4);
    float*  e0  = (float*)carve((size_t)6400 * Dk * 4);
    float*  e1  = (float*)carve((size_t)25600 * Dk * 4);
    float*  e2  = (float*)carve((size_t)102400 * Dk * 4);
    __bf16* Xb  = (__bf16*)carve((size_t)102400 * Dk * 2);
    __bf16* Wb  = (__bf16*)carve((size_t)4 * Dk * Dk * 2);
    float*  rec = (float*)carve((size_t)6400 * Dk * 4);
    float*  shb = (float*)carve((size_t)Bk * Sk * Dk * 4);

    auto blks = [](int n) { return (n + 255) / 256; };

    // phase 1: indices + initial gathers + bf16 weights
    k_indices   <<<blks(102400), 256, 0, stream>>>(question, q_nb, s_nb, n1, n2);
    k_gather_e0 <<<blks(6400 * Dk),  256, 0, stream>>>(question, embq, e0);
    k_gather_e1 <<<blks(25600 * Dk), 256, 0, stream>>>(n1, embc, e1);
    k_wbf       <<<blks(4 * Dk * Dk), 256, 0, stream>>>(agg_w, agg_last_w, Wb);

    // i=0: j=0 (reads old e1), j=1 (reads gathered e2_init), j=2 (on-the-fly n3)
    k_prep_generic<<<blks(6400 * Dk), 256, 0, stream>>>(e0, e1, Xb, 6400);
    k_gemm_tanh_bf16<<<6400 / 128, 256, 0, stream>>>(Xb, Wb + 0 * Dk * Dk, agg_b + 0 * Dk, e0, 6400);
    k_prep_x1<<<blks(25600 * Dk), 256, 0, stream>>>(e1, n2, embq, Xb);
    k_gemm_tanh_bf16<<<25600 / 128, 256, 0, stream>>>(Xb, Wb + 1 * Dk * Dk, agg_b + 1 * Dk, e1, 25600);
    k_prep_x2<<<blks(102400 * Dk), 256, 0, stream>>>(n2, q_nb, embq, embc, Xb);
    k_gemm_tanh_bf16<<<102400 / 128, 256, 0, stream>>>(Xb, Wb + 2 * Dk * Dk, agg_b + 2 * Dk, e2, 102400);
    // i=1: j=0, j=1
    k_prep_generic<<<blks(6400 * Dk), 256, 0, stream>>>(e0, e1, Xb, 6400);
    k_gemm_tanh_bf16<<<6400 / 128, 256, 0, stream>>>(Xb, Wb + 0 * Dk * Dk, agg_b + 0 * Dk, e0, 6400);
    k_prep_generic<<<blks(25600 * Dk), 256, 0, stream>>>(e1, e2, Xb, 25600);
    k_gemm_tanh_bf16<<<25600 / 128, 256, 0, stream>>>(Xb, Wb + 1 * Dk * Dk, agg_b + 1 * Dk, e1, 25600);
    // i=2: j=0
    k_prep_generic<<<blks(6400 * Dk), 256, 0, stream>>>(e0, e1, Xb, 6400);
    k_gemm_tanh_bf16<<<6400 / 128, 256, 0, stream>>>(Xb, Wb + 0 * Dk * Dk, agg_b + 0 * Dk, e0, 6400);
    // agg = tanh(e0 @ agg_last_w^T + b)
    k_copy_bf<<<blks(6400 * Dk), 256, 0, stream>>>(e0, Xb, 6400 * Dk);
    k_gemm_tanh_bf16<<<6400 / 128, 256, 0, stream>>>(Xb, Wb + 3 * Dk * Dk, agg_last_b, e0, 6400);
    k_emb_q_rec<<<blks(6400 * Dk), 256, 0, stream>>>(mask, question, e0, embq, rec);

    // zero y_hat (col 1 must be 0) and the sh state buffer (ws is poisoned)
    k_zero<<<blks(Bk * Sk * Dk), 256, 0, stream>>>(yhat, Bk * Sk, shb, Bk * Sk * Dk);

    // phase 2: sequential scan, one workgroup, 16 waves, ~293KB LDS
    const size_t smem_bytes =
        (4096 + 4096 + 8192 + 12288 + 12288 + 4096 + 6400 + 20480 + 1760 + 352 + 160) * 4
        + (320 + 352) * 4;
    static bool attr_done = false;
    if (!attr_done) {  // host-side attribute set; not a stream op (capture-safe)
        hipFuncSetAttribute((const void*)k_scan,
                            hipFuncAttributeMaxDynamicSharedMemorySize, (int)smem_bytes);
        attr_done = true;
    }
    k_scan<<<1, 512, smem_bytes, stream>>>(
        rec, shb, question, response, qc_idx, qc_mask, embq, embc, embcorr,
        w1ih, w1hh, b1ih, b1hh, w2ih, w2hh, b2ih, b2hh,
        query_w, query_b, key_w, key_b, mlpW_w, mlpW_b, h1_init, h2_init, yhat);
}
